// CausalSelfAttention_62371515072642
// MI455X (gfx1250) — matrix operand order repaired and checked by reading, not compiled, not run on hardware
//
#include <hip/hip_runtime.h>

// ---------------- constants for this problem ----------------
#define B_  4
#define T_  2048
#define C_  1024
#define H_  16
#define D_  64
#define BH_ (B_ * H_)   // 64

typedef __bf16 v16bf __attribute__((ext_vector_type(16)));
typedef __bf16 v8bf  __attribute__((ext_vector_type(8)));
typedef float  v8f   __attribute__((ext_vector_type(8)));

static __device__ __forceinline__ v8f zero8() {
    v8f z;
#pragma unroll
    for (int i = 0; i < 8; ++i) z[i] = 0.0f;
    return z;
}

static __device__ __forceinline__ v16bf combine16(v8bf lo, v8bf hi) {
    v16bf r;
#pragma unroll
    for (int i = 0; i < 8; ++i) { r[i] = lo[i]; r[8 + i] = hi[i]; }
    return r;
}

// A-operand fragment (16-bit, 16x32 MxK): lane half h holds
// k = kbase + h*16 + {0..15}. Two 16B loads.
static __device__ __forceinline__ v16bf load_a16(const __bf16* rowp, int kbase, int h) {
    v8bf lo = *(const v8bf*)(rowp + kbase + h * 16);
    v8bf hi = *(const v8bf*)(rowp + kbase + h * 16 + 8);
    return combine16(lo, hi);
}

// B-operand fragment (16-bit, 32x16 KxN): lane half h covers k = kbase + h*16 + i.
static __device__ __forceinline__ v16bf load_b16(const __bf16* colp, int kbase, int h) {
    v8bf lo = *(const v8bf*)(colp + kbase + h * 16);
    v8bf hi = *(const v8bf*)(colp + kbase + h * 16 + 8);
    return combine16(lo, hi);
}

static __device__ __forceinline__ v8f wmma_bf16(v16bf a, v16bf b, v8f c) {
    return __builtin_amdgcn_wmma_f32_16x16x32_bf16(
        false, a, false, b, (short)0, c, false, false);
}

// ---------------- kernel 1: fp32 -> bf16 convert ----------------
__global__ void __launch_bounds__(256) cvt_bf16_kernel(const float* __restrict__ in,
                                                       __bf16* __restrict__ out, int n) {
    int i = blockIdx.x * blockDim.x + threadIdx.x;
    int stride = gridDim.x * blockDim.x;
    for (; i < n; i += stride) out[i] = (__bf16)in[i];
}

// ---------------- kernel 2: fused QKV projection, 64x64 tile per wave -------
// Tiles: 128 row-blocks (B*T/64) x 48 col-blocks (3*C/64). 16 WMMAs per k-step
// from 4 A-frags + 4 B-frags (0.5 KB/WMMA of fragment traffic).
// Writes Q (pre-scaled 1/8) and K as [B,H,T,D] bf16, V transposed [B,H,D,T].
__global__ void __launch_bounds__(256) qkv_kernel(const __bf16* __restrict__ xb,
                                                  const __bf16* __restrict__ wb,
                                                  __bf16* __restrict__ Qb,
                                                  __bf16* __restrict__ Kb,
                                                  __bf16* __restrict__ Vt) {
    const int lane = threadIdx.x & 31;
    const int h = lane >> 4, ln = lane & 15;
    const int wid = blockIdx.x * (blockDim.x >> 5) + (threadIdx.x >> 5);
    const int tileN = wid % 48;
    const int tileM = wid / 48;
    const int gm = tileM * 64, gn = tileN * 64;

    const int sel = gn >> 10;                 // 0=Q 1=K 2=V (64 | 1024 -> uniform)
    const int cbase = gn & 1023;

    const __bf16* arow[4];
    const __bf16* brow[4];
#pragma unroll
    for (int i = 0; i < 4; ++i) {
        arow[i] = xb + (size_t)(gm + i * 16 + ln) * C_;
        brow[i] = wb + (size_t)sel * C_ * C_ + (size_t)(cbase + i * 16 + ln) * C_;
    }

    v8f acc[4][4];
#pragma unroll
    for (int i = 0; i < 4; ++i)
#pragma unroll
        for (int j = 0; j < 4; ++j) acc[i][j] = zero8();

    for (int kb = 0; kb < C_; kb += 32) {
        v16bf a[4], b[4];
#pragma unroll
        for (int i = 0; i < 4; ++i) a[i] = load_a16(arow[i], kb, h);
#pragma unroll
        for (int j = 0; j < 4; ++j) b[j] = load_b16(brow[j], kb, h);
#pragma unroll
        for (int i = 0; i < 4; ++i)
#pragma unroll
            for (int j = 0; j < 4; ++j)
                acc[i][j] = wmma_bf16(a[i], b[j], acc[i][j]);
    }

#pragma unroll
    for (int j = 0; j < 4; ++j) {
        const int co = cbase + j * 16 + ln;    // out channel within matrix
        const int head = co >> 6, d = co & 63;
#pragma unroll
        for (int i = 0; i < 4; ++i) {
#pragma unroll
            for (int r = 0; r < 8; ++r) {
                const int gr = gm + i * 16 + r + 8 * h;   // row in [0, B*T)
                const int b_ = gr >> 11, t = gr & (T_ - 1);
                const float v = acc[i][j][r];
                const size_t bh = (size_t)(b_ * H_ + head);
                if (sel == 0)      Qb[(bh * T_ + t) * D_ + d] = (__bf16)(v * 0.125f);
                else if (sel == 1) Kb[(bh * T_ + t) * D_ + d] = (__bf16)v;
                else               Vt[(bh * D_ + d) * T_ + t] = (__bf16)v;
            }
        }
    }
}

// ---------------- kernel 3: flash attention, 32 queries per wave ------------
// S^T = K * Q^T (lane = query column) so row-softmax is 8 regs + one
// shfl_xor(16) across wave halves; K/V fragments reused across both q-tiles.
__global__ void __launch_bounds__(256) attn_kernel(const __bf16* __restrict__ Qb,
                                                   const __bf16* __restrict__ Kb,
                                                   const __bf16* __restrict__ Vt,
                                                   __bf16* __restrict__ Y) {
    const int lane = threadIdx.x & 31;
    const int h = lane >> 4, ln = lane & 15;
    const int wid = blockIdx.x * (blockDim.x >> 5) + (threadIdx.x >> 5);
    const int bh = wid >> 6;                 // T/32 = 64 q-blocks per (b,h)
    const int q0 = (wid & 63) << 5;          // 32 queries per wave

    v16bf bq[2][2];                          // [q-tile][d-chunk]
#pragma unroll
    for (int qt = 0; qt < 2; ++qt) {
        const __bf16* qrow = Qb + (size_t)(bh * T_ + q0 + qt * 16 + ln) * D_;
        bq[qt][0] = load_b16(qrow, 0, h);
        bq[qt][1] = load_b16(qrow, 32, h);
    }

    v8f acc[2][4];
#pragma unroll
    for (int qt = 0; qt < 2; ++qt)
#pragma unroll
        for (int j = 0; j < 4; ++j) acc[qt][j] = zero8();

    float mrow[2] = {-3.0e38f, -3.0e38f};
    float lrow[2] = {0.0f, 0.0f};
    const int qidx[2] = {q0 + ln, q0 + 16 + ln};

    const __bf16* vbase = Vt + (size_t)(bh * D_ + ln) * T_;

    for (int kb = 0; kb < q0 + 32; kb += 32) {
        // K fragments for keys kb..kb+31 (shared by both q-tiles)
        v16bf aK[2][2];
#pragma unroll
        for (int tt = 0; tt < 2; ++tt) {
            const __bf16* krow = Kb + (size_t)(bh * T_ + kb + tt * 16 + ln) * D_;
            aK[tt][0] = load_a16(krow, 0, h);
            aK[tt][1] = load_a16(krow, 32, h);
        }

        v16bf pb[2];
#pragma unroll
        for (int qt = 0; qt < 2; ++qt) {
            v8f s0 = zero8(), s1 = zero8();
            s0 = wmma_bf16(aK[0][0], bq[qt][0], s0);
            s0 = wmma_bf16(aK[0][1], bq[qt][1], s0);
            s1 = wmma_bf16(aK[1][0], bq[qt][0], s1);
            s1 = wmma_bf16(aK[1][1], bq[qt][1], s1);

            // causal mask + local max (lane holds keys kb+8h+r, kb+16+8h+r)
            float p0[8], p1[8];
            float mloc = -3.0e38f;
#pragma unroll
            for (int r = 0; r < 8; ++r) {
                const int k0 = kb + 8 * h + r;
                const int k1 = k0 + 16;
                p0[r] = (k0 <= qidx[qt]) ? s0[r] : -3.0e38f;
                p1[r] = (k1 <= qidx[qt]) ? s1[r] : -3.0e38f;
                mloc = fmaxf(mloc, fmaxf(p0[r], p1[r]));
            }
            mloc = fmaxf(mloc, __shfl_xor(mloc, 16, 32));
            const float mnew = fmaxf(mrow[qt], mloc);
            const float alpha = __expf(mrow[qt] - mnew);

            float ssum = 0.0f;
#pragma unroll
            for (int r = 0; r < 8; ++r) {
                p0[r] = __expf(p0[r] - mnew);
                p1[r] = __expf(p1[r] - mnew);
                ssum += p0[r] + p1[r];
            }
            ssum += __shfl_xor(ssum, 16, 32);
            lrow[qt] = lrow[qt] * alpha + ssum;
            mrow[qt] = mnew;

#pragma unroll
            for (int j = 0; j < 4; ++j)
#pragma unroll
                for (int r = 0; r < 8; ++r) acc[qt][j][r] *= alpha;

            // Repack P^T (S^T D-layout) into B-fragment layout via one
            // cross-half shuffle per register.
#pragma unroll
            for (int r = 0; r < 8; ++r) {
                const float o0 = __shfl_xor(p0[r], 16, 32);
                const float o1 = __shfl_xor(p1[r], 16, 32);
                pb[qt][r]     = (__bf16)((h == 0) ? p0[r] : o1);
                pb[qt][8 + r] = (__bf16)((h == 0) ? o0    : p1[r]);
            }
        }

        // V fragments (shared by both q-tiles): y^T += V^T * P^T
        v16bf aV[4];
#pragma unroll
        for (int j = 0; j < 4; ++j) aV[j] = load_a16(vbase + j * 16 * T_, kb, h);
#pragma unroll
        for (int qt = 0; qt < 2; ++qt)
#pragma unroll
            for (int j = 0; j < 4; ++j)
                acc[qt][j] = wmma_bf16(aV[j], pb[qt], acc[qt][j]);
    }

    // epilogue: normalize, store as [B,T,H*D] bf16 (lane = query column)
    const int b = bh >> 4, head = bh & 15;
#pragma unroll
    for (int qt = 0; qt < 2; ++qt) {
        const float invl = 1.0f / lrow[qt];
        __bf16* yrow = Y + ((size_t)(b * T_ + qidx[qt]) * H_ + head) * D_;
#pragma unroll
        for (int j = 0; j < 4; ++j)
#pragma unroll
            for (int r = 0; r < 8; ++r)
                yrow[j * 16 + r + 8 * h] = (__bf16)(acc[qt][j][r] * invl);
    }
}

// ---------------- kernel 4: output projection, 64x64 tile per wave ----------
__global__ void __launch_bounds__(256) proj_kernel(const __bf16* __restrict__ yb,
                                                   const __bf16* __restrict__ wp,
                                                   float* __restrict__ out) {
    const int lane = threadIdx.x & 31;
    const int h = lane >> 4, ln = lane & 15;
    const int wid = blockIdx.x * (blockDim.x >> 5) + (threadIdx.x >> 5);
    const int tileN = wid & 15;              // C/64 = 16 col blocks
    const int tileM = wid >> 4;
    const int gm = tileM * 64, gn = tileN * 64;

    const __bf16* arow[4];
    const __bf16* brow[4];
#pragma unroll
    for (int i = 0; i < 4; ++i) {
        arow[i] = yb + (size_t)(gm + i * 16 + ln) * C_;
        brow[i] = wp + (size_t)(gn + i * 16 + ln) * C_;
    }

    v8f acc[4][4];
#pragma unroll
    for (int i = 0; i < 4; ++i)
#pragma unroll
        for (int j = 0; j < 4; ++j) acc[i][j] = zero8();

    for (int kb = 0; kb < C_; kb += 32) {
        v16bf a[4], b[4];
#pragma unroll
        for (int i = 0; i < 4; ++i) a[i] = load_a16(arow[i], kb, h);
#pragma unroll
        for (int j = 0; j < 4; ++j) b[j] = load_b16(brow[j], kb, h);
#pragma unroll
        for (int i = 0; i < 4; ++i)
#pragma unroll
            for (int j = 0; j < 4; ++j)
                acc[i][j] = wmma_bf16(a[i], b[j], acc[i][j]);
    }

#pragma unroll
    for (int i = 0; i < 4; ++i)
#pragma unroll
        for (int j = 0; j < 4; ++j)
#pragma unroll
            for (int r = 0; r < 8; ++r) {
                const int m = gm + i * 16 + r + 8 * h;
                out[(size_t)m * C_ + gn + j * 16 + ln] = acc[i][j][r];
            }
}

// ---------------- launch ----------------
extern "C" void kernel_launch(void* const* d_in, const int* in_sizes, int n_in,
                              void* d_out, int out_size, void* d_ws, size_t ws_size,
                              hipStream_t stream) {
    const float* x  = (const float*)d_in[0];
    const float* Wq = (const float*)d_in[1];
    const float* Wk = (const float*)d_in[2];
    const float* Wv = (const float*)d_in[3];
    const float* Wp = (const float*)d_in[4];

    const int nX = B_ * T_ * C_;      // 8388608
    const int nW = C_ * C_;           // 1048576

    // workspace layout (bf16), 16B-aligned chunks; total 88 MB
    char* ws = (char*)d_ws;
    __bf16* xb = (__bf16*)(ws);                             // 16 MB
    __bf16* wb = (__bf16*)(ws + (size_t)16 * 1024 * 1024);  //  8 MB (Wq,Wk,Wv,Wp)
    __bf16* Qb = (__bf16*)(ws + (size_t)24 * 1024 * 1024);  // 16 MB [B,H,T,D]
    __bf16* Kb = (__bf16*)(ws + (size_t)40 * 1024 * 1024);  // 16 MB [B,H,T,D]
    __bf16* Vt = (__bf16*)(ws + (size_t)56 * 1024 * 1024);  // 16 MB [B,H,D,T]
    __bf16* Yb = (__bf16*)(ws + (size_t)72 * 1024 * 1024);  // 16 MB [B,T,H*D]

    cvt_bf16_kernel<<<2048, 256, 0, stream>>>(x,  xb,                 nX);
    cvt_bf16_kernel<<<512,  256, 0, stream>>>(Wq, wb + 0 * (size_t)nW, nW);
    cvt_bf16_kernel<<<512,  256, 0, stream>>>(Wk, wb + 1 * (size_t)nW, nW);
    cvt_bf16_kernel<<<512,  256, 0, stream>>>(Wv, wb + 2 * (size_t)nW, nW);
    cvt_bf16_kernel<<<512,  256, 0, stream>>>(Wp, wb + 3 * (size_t)nW, nW);

    // QKV: 128 * 48 = 6144 wave-tiles, 8 waves/block -> 768 blocks
    qkv_kernel<<<768, 256, 0, stream>>>(xb, wb, Qb, Kb, Vt);

    // attention: 64 * 64 = 4096 wave-tiles -> 512 blocks
    attn_kernel<<<512, 256, 0, stream>>>(Qb, Kb, Vt, Yb);

    // output projection: 128 * 16 = 2048 wave-tiles -> 256 blocks
    proj_kernel<<<256, 256, 0, stream>>>(Yb, wb + 3 * (size_t)nW, (float*)d_out);
}
